// LabelSmoothLoss_84868553769049
// MI455X (gfx1250) — compile-verified, matrix operand order checked by reading, not run on hardware
//
#include <hip/hip_runtime.h>
#include <hip/hip_bf16.h>
#include <math.h>

typedef _Float16 half8 __attribute__((ext_vector_type(8)));
typedef _Float16 v16h  __attribute__((ext_vector_type(16)));
typedef float    v8f   __attribute__((ext_vector_type(8)));

#define BM 128
#define BN 128
#define BK 32
#define STR 40   // LDS row stride in halves: 80 bytes, multiple of 16B

// ---------------------------------------------------------------------------
// prep kernels
// ---------------------------------------------------------------------------

// One wave per row of predicts: write Aconcat = [Ep | E] (f16), sumE per row.
__global__ __launch_bounds__(256)
void prep_pred(const float* __restrict__ pred, _Float16* __restrict__ Ac,
               float* __restrict__ sumE, int B, int L) {
    int wid  = threadIdx.x >> 5;
    int lane = threadIdx.x & 31;
    int b = blockIdx.x * 8 + wid;
    if (b >= B) return;
    const float thr = 0.5f + 0.1f;   // PREDICT_GAP
    float s = 0.0f;
    size_t rowA = (size_t)b * (2 * (size_t)L);
    for (int l = lane; l < L; l += 32) {
        float p = pred[(size_t)b * L + l];
        float e = (p > thr) ? 1.0f : 0.0f;
        Ac[rowA + l]     = (_Float16)(e * p);   // Ep
        Ac[rowA + L + l] = (_Float16)e;         // E
        s += e;
    }
    for (int off = 16; off > 0; off >>= 1) s += __shfl_down(s, off, 32);
    if (lane == 0) sumE[b] = s;
}

// Build BcT = [M1 ; M2]^T  (f16, [L rows][2L cols]):
//   BcT[j][i]   = M1[i][j] = incr*a - decr*(1-a)
//   BcT[j][L+i] = M2[i][j] = decr
// Stored transposed so the GEMM stages it with pure b128 copies.
__global__ __launch_bounds__(256)
void prep_adj(const float* __restrict__ adj, const float* __restrict__ lpv,
              _Float16* __restrict__ BcT, int L) {
    size_t idx = (size_t)blockIdx.x * blockDim.x + threadIdx.x;
    size_t total = (size_t)L * L;
    if (idx >= total) return;
    int i = (int)(idx / L);
    int j = (int)(idx % L);
    float a = adj[idx];
    float p = lpv[j];
    bool diag = (i == j);
    bool incr = !diag && (a > p + 0.05f);   // RELATION_GAP
    bool decr = !diag && (a < p - 0.05f);
    float m1 = incr ? a : (decr ? (a - 1.0f) : 0.0f);
    size_t rowT = (size_t)j * (2 * (size_t)L);
    BcT[rowT + i]     = (_Float16)m1;
    BcT[rowT + L + i] = (_Float16)(decr ? 1.0f : 0.0f);
}

// One wave per row of similarities: f16 copy, row sum, zero rowacc.
__global__ __launch_bounds__(256)
void prep_sim(const float* __restrict__ sim, _Float16* __restrict__ Sh,
              float* __restrict__ simrowsum, float* __restrict__ rowacc, int B) {
    int wid  = threadIdx.x >> 5;
    int lane = threadIdx.x & 31;
    int b = blockIdx.x * 8 + wid;
    if (b >= B) return;
    float s = 0.0f;
    for (int j = lane; j < B; j += 32) {
        float v = sim[(size_t)b * B + j];
        Sh[(size_t)b * B + j] = (_Float16)v;
        s += v;
    }
    for (int off = 16; off > 0; off >>= 1) s += __shfl_down(s, off, 32);
    if (lane == 0) { simrowsum[b] = s; rowacc[b] = 0.0f; }
}

// ---------------------------------------------------------------------------
// WMMA GEMM: D = A @ Bt^T with A [M][K], Bt [N][K], both row-major f16.
// Block tile 128x128, 8 waves (4M x 2N), wave tile 32x64 = 2x4 WMMA tiles.
// Fragment layout per CDNA5 ISA 16-bit 16x32: lanes 0-15 hold K{0..7,16..23},
// lanes 16-31 hold K{8..15,24..31} of their row -> two contiguous b128 loads.
// ---------------------------------------------------------------------------

__device__ inline v16h load_frag(const _Float16* lds, int rowBase, int lane) {
    int r  = lane & 15;
    int kb = (lane >> 4) << 3;   // 0 or 8
    const _Float16* p = lds + (size_t)(rowBase + r) * STR;
    half8 lo = *(const half8*)(p + kb);
    half8 hi = *(const half8*)(p + kb + 16);
    v16h f;
#pragma unroll
    for (int i = 0; i < 8; ++i) { f[i] = lo[i]; f[i + 8] = hi[i]; }
    return f;
}

// MODE 0: candidates epilogue   CdT[col][row] = (pred + acc) / (1 + sumE - E)
// MODE 1: residual epilogue     rowacc[row] += sum_col (pred - acc)^2
template <int MODE>
__global__ __launch_bounds__(256)
void gemm_wmma(const _Float16* __restrict__ A, const _Float16* __restrict__ Bt,
               const float* __restrict__ pred, const float* __restrict__ sumE,
               _Float16* __restrict__ CdT, float* __restrict__ rowacc,
               int M, int N, int K) {
    __shared__ _Float16 ldsA[BM * STR];
    __shared__ _Float16 ldsB[BN * STR];

    const int tid   = threadIdx.x;
    const int lane  = tid & 31;
    const int wid   = tid >> 5;
    const int waveM = wid & 3;     // 0..3  -> 32-row band
    const int waveN = wid >> 2;    // 0..1  -> 64-col band
    const int rowBlk = blockIdx.y * BM;
    const int colBlk = blockIdx.x * BN;

    v8f acc[2][4] = {};

    const int srow = tid >> 1;           // 0..127 (tile row for staging)
    const int sseg = (tid & 1) << 4;     // 0 or 16 halves

    const _Float16* ga = A  + (size_t)(rowBlk + srow) * K + sseg;
    const _Float16* gb = Bt + (size_t)(colBlk + srow) * K + sseg;
    _Float16* la = ldsA + (size_t)srow * STR + sseg;
    _Float16* lb = ldsB + (size_t)srow * STR + sseg;

    for (int kb = 0; kb < K; kb += BK) {
        // stage A and B^T tiles: pure 16B vector copies, no scalar LDS stores
        half8 a0 = *(const half8*)(ga + kb);
        half8 a1 = *(const half8*)(ga + kb + 8);
        half8 b0 = *(const half8*)(gb + kb);
        half8 b1 = *(const half8*)(gb + kb + 8);
        *(half8*)(la)     = a0;
        *(half8*)(la + 8) = a1;
        *(half8*)(lb)     = b0;
        *(half8*)(lb + 8) = b1;
        if (kb + BK < K) {   // keep next K tile warm in near caches
            __builtin_prefetch(ga + kb + BK, 0, 3);
            __builtin_prefetch(gb + kb + BK, 0, 3);
        }
        __syncthreads();

        v16h af[2], bf[4];
#pragma unroll
        for (int mi = 0; mi < 2; ++mi)
            af[mi] = load_frag(ldsA, waveM * 32 + mi * 16, lane);
#pragma unroll
        for (int ni = 0; ni < 4; ++ni)
            bf[ni] = load_frag(ldsB, waveN * 64 + ni * 16, lane);

#pragma unroll
        for (int mi = 0; mi < 2; ++mi)
#pragma unroll
            for (int ni = 0; ni < 4; ++ni)
                acc[mi][ni] = __builtin_amdgcn_wmma_f32_16x16x32_f16(
                    false, af[mi], false, bf[ni], (short)0, acc[mi][ni], false, false);
        __syncthreads();
    }

    // Epilogue. C/D layout: VGPR r -> row r (lanes 0-15) / row 8+r (lanes 16-31),
    // column = lane & 15.
    const int nLane   = lane & 15;
    const int rowHalf = (lane >> 4) * 8;
    const float thr = 0.5f + 0.1f;
#pragma unroll
    for (int mi = 0; mi < 2; ++mi) {
#pragma unroll
        for (int r = 0; r < 8; ++r) {
            int row = rowBlk + waveM * 32 + mi * 16 + rowHalf + r;
            if (MODE == 0) {
                float se = sumE[row];
#pragma unroll
                for (int ni = 0; ni < 4; ++ni) {
                    int col = colBlk + waveN * 64 + ni * 16 + nLane;
                    float p = pred[(size_t)row * N + col];
                    float e = (p > thr) ? 1.0f : 0.0f;
                    float rel = 1.0f + se - e;
                    // store candidates transposed: CdT [N][M]
                    CdT[(size_t)col * M + row] = (_Float16)((p + acc[mi][ni][r]) / rel);
                }
            } else {
                float v = 0.0f;
#pragma unroll
                for (int ni = 0; ni < 4; ++ni) {
                    int col = colBlk + waveN * 64 + ni * 16 + nLane;
                    float p = pred[(size_t)row * N + col];
                    float d = p - acc[mi][ni][r];
                    v += d * d;
                }
                // reduce over the 16 lanes sharing this row, then one atomic
                for (int m = 8; m > 0; m >>= 1) v += __shfl_xor(v, m, 32);
                if ((lane & 15) == 0) atomicAdd(&rowacc[row], v);
            }
        }
    }
}

// ---------------------------------------------------------------------------
// final masked-mean reduction -> scalar
// ---------------------------------------------------------------------------
__global__ __launch_bounds__(256)
void final_reduce(const float* __restrict__ rowacc, const float* __restrict__ simrowsum,
                  float* __restrict__ out, int B) {
    __shared__ float st[256];
    __shared__ float sc[256];
    float t = 0.0f, c = 0.0f;
    for (int b = threadIdx.x; b < B; b += 256) {
        if (simrowsum[b] != 0.0f) { t += sqrtf(rowacc[b]); c += 1.0f; }
    }
    st[threadIdx.x] = t; sc[threadIdx.x] = c;
    __syncthreads();
    for (int off = 128; off > 0; off >>= 1) {
        if (threadIdx.x < off) {
            st[threadIdx.x] += st[threadIdx.x + off];
            sc[threadIdx.x] += sc[threadIdx.x + off];
        }
        __syncthreads();
    }
    if (threadIdx.x == 0) {
        float cnt = sc[0];
        out[0] = (cnt == 0.0f) ? 0.0f : st[0] / fmaxf(cnt, 1.0f);
    }
}

// ---------------------------------------------------------------------------
extern "C" void kernel_launch(void* const* d_in, const int* in_sizes, int n_in,
                              void* d_out, int out_size, void* d_ws, size_t ws_size,
                              hipStream_t stream) {
    const float* predicts = (const float*)d_in[0];   // [B, L]
    const float* sims     = (const float*)d_in[1];   // [B, B]
    const float* adjList  = (const float*)d_in[2];   // [L, L]
    const float* lpv      = (const float*)d_in[3];   // [L]
    float* out = (float*)d_out;

    const int L = in_sizes[3];
    const int B = in_sizes[0] / L;

    char* ws = (char*)d_ws;
    size_t off = 0;
    _Float16* Ac  = (_Float16*)(ws + off); off += (size_t)B * 2 * L * sizeof(_Float16); // [B, 2L]
    _Float16* BcT = (_Float16*)(ws + off); off += (size_t)L * 2 * L * sizeof(_Float16); // [L, 2L]
    _Float16* Sh  = (_Float16*)(ws + off); off += (size_t)B * B * sizeof(_Float16);     // [B, B]
    _Float16* CdT = (_Float16*)(ws + off); off += (size_t)L * B * sizeof(_Float16);     // [L, B]
    float* sumE   = (float*)(ws + off);    off += (size_t)B * sizeof(float);
    float* simrs  = (float*)(ws + off);    off += (size_t)B * sizeof(float);
    float* rowacc = (float*)(ws + off);    off += (size_t)B * sizeof(float);

    // prep
    prep_pred<<<(B + 7) / 8, 256, 0, stream>>>(predicts, Ac, sumE, B, L);
    {
        size_t total = (size_t)L * L;
        prep_adj<<<(unsigned)((total + 255) / 256), 256, 0, stream>>>(adjList, lpv, BcT, L);
    }
    prep_sim<<<(B + 7) / 8, 256, 0, stream>>>(sims, Sh, simrs, rowacc, B);

    // GEMM1: candidates^T = ((predicts + [Ep|E] @ [M1;M2]) / rel)^T  (M=B, N=L, K=2L)
    dim3 g1(L / BN, B / BM);
    gemm_wmma<0><<<g1, 256, 0, stream>>>(Ac, BcT, predicts, sumE, CdT, rowacc, B, L, 2 * L);

    // GEMM2: resid = predicts - Sh @ candidates ; rowacc += resid^2  (M=B, N=L, K=B)
    dim3 g2(L / BN, B / BM);
    gemm_wmma<1><<<g2, 256, 0, stream>>>(Sh, CdT, predicts, sumE, CdT, rowacc, B, L, B);

    // masked mean of row norms
    final_reduce<<<1, 256, 0, stream>>>(rowacc, simrs, out, B);
}
